// GNN_64836826301100
// MI455X (gfx1250) — compile-verified, matrix operand order checked by reading, not run on hardware
//
#include <hip/hip_runtime.h>
#include <hip/hip_bf16.h>

#define N_NODES 100000
#define N_EDGES 1600000
#define IN_C 128
#define HID_C 64
#define OUT_C 32

typedef __attribute__((ext_vector_type(16))) __bf16 v16bf;
typedef __attribute__((ext_vector_type(8)))  float  v8f;

// ---------------------------------------------------------------------------
// Degree / norm kernels
// ---------------------------------------------------------------------------
__global__ void init_deg_kernel(float* __restrict__ deg, int n) {
    int i = blockIdx.x * blockDim.x + threadIdx.x;
    if (i < n) deg[i] = 1.0f;                 // self-loop contributes 1
}

__global__ void count_deg_kernel(const long long* __restrict__ col,
                                 float* __restrict__ deg, int e) {
    int i = blockIdx.x * blockDim.x + threadIdx.x;
    if (i < e) atomicAdd(&deg[(int)col[i]], 1.0f);
}

__global__ void rsqrt_kernel(float* __restrict__ deg, int n) {
    int i = blockIdx.x * blockDim.x + threadIdx.x;
    if (i < n) deg[i] = __frsqrt_rn(deg[i]);  // deg >= 1 always
}

// ---------------------------------------------------------------------------
// WMMA GEMM:  C[M x N] = A[M x K] * B[K x N], A/B f32 row-major -> bf16,
// f32 accumulate.  One wave32 computes one 16x16 tile.  M % 16 == 0.
// ---------------------------------------------------------------------------
template<int K, int N>
__global__ void gemm_bf16_wmma(const float* __restrict__ A,
                               const float* __restrict__ B,
                               float* __restrict__ C, int M) {
    const int waveId = (blockIdx.x * blockDim.x + threadIdx.x) >> 5;
    const int lane   = threadIdx.x & 31;
    constexpr int tilesN = N / 16;
    const int mTile = waveId / tilesN;
    const int nTile = waveId % tilesN;
    if (mTile * 16 >= M) return;              // wave-uniform: EXEC stays all-1s

    const int aRow  = mTile * 16 + (lane & 15);
    const int bCol  = nTile * 16 + (lane & 15);
    const int kbase = (lane >> 4) * 8;        // 0 for lanes 0-15, 8 for 16-31

    v8f acc = {};
    for (int k0 = 0; k0 < K; k0 += 32) {
        v16bf a, b;
        const float* ap = A + (size_t)aRow * K + k0 + kbase;
#pragma unroll
        for (int i = 0; i < 8; ++i) {
            a[i]     = (__bf16)ap[i];          // K = k0+kbase+i
            a[i + 8] = (__bf16)ap[16 + i];     // K = k0+kbase+16+i
        }
#pragma unroll
        for (int i = 0; i < 8; ++i) {
            b[i]     = (__bf16)B[(size_t)(k0 + kbase + i)      * N + bCol];
            b[i + 8] = (__bf16)B[(size_t)(k0 + kbase + 16 + i) * N + bCol];
        }
        acc = __builtin_amdgcn_wmma_f32_16x16x32_bf16(
            /*neg_a=*/false, a, /*neg_b=*/false, b,
            /*c_mod=*/(short)0, acc, /*reuse_a=*/false, /*reuse_b=*/false);
    }

    const int outRow0 = mTile * 16 + ((lane >> 4) * 8);
    const int outCol  = nTile * 16 + (lane & 15);
#pragma unroll
    for (int v = 0; v < 8; ++v)
        C[(size_t)(outRow0 + v) * N + outCol] = acc[v];
}

// ---------------------------------------------------------------------------
// Aggregation:  agg = b + sum_{edges} norm * h[row]  (+ self-loop term)
// ---------------------------------------------------------------------------
template<int C>
__global__ void init_agg_kernel(const float* __restrict__ hw,
                                const float* __restrict__ dinv,
                                const float* __restrict__ bias,
                                float* __restrict__ agg, int n) {
    int t = blockIdx.x * blockDim.x + threadIdx.x;
    if (t >= n * C) return;
    int i = t / C, c = t % C;
    float di = dinv[i];
    agg[t] = bias[c] + hw[t] * di * di;       // self-loop: norm = dinv[i]^2
}

template<int C>
__global__ void edge_agg_kernel(const float* __restrict__ h,
                                const long long* __restrict__ rows,
                                const long long* __restrict__ cols,
                                const float* __restrict__ dinv,
                                float* __restrict__ agg, long long e) {
    long long t = (long long)blockIdx.x * blockDim.x + threadIdx.x;
    if (t >= e * C) return;
    long long ed = t / C;
    int c = (int)(t % C);
    int r  = (int)rows[ed];
    int cl = (int)cols[ed];
    float nrm = dinv[r] * dinv[cl];
    atomicAdd(&agg[(size_t)cl * C + c], h[(size_t)r * C + c] * nrm);
}

__global__ void relu_kernel(float* __restrict__ x, int n) {
    int i = blockIdx.x * blockDim.x + threadIdx.x;
    if (i < n) x[i] = fmaxf(x[i], 0.0f);
}

// ---------------------------------------------------------------------------
// Launch
// ---------------------------------------------------------------------------
extern "C" void kernel_launch(void* const* d_in, const int* in_sizes, int n_in,
                              void* d_out, int out_size, void* d_ws, size_t ws_size,
                              hipStream_t stream) {
    const float*     x    = (const float*)d_in[0];
    const long long* eidx = (const long long*)d_in[1];   // int64, shape (2, E)
    const float*     W1   = (const float*)d_in[2];
    const float*     b1   = (const float*)d_in[3];
    const float*     W2   = (const float*)d_in[4];
    const float*     b2   = (const float*)d_in[5];
    float*           out  = (float*)d_out;

    const long long* rows = eidx;
    const long long* cols = eidx + N_EDGES;

    // Workspace layout (bytes)
    char* ws = (char*)d_ws;
    float* dinv = (float*)(ws);                               // 0.4 MB (deg -> dinv in place)
    float* xw   = (float*)(ws + (size_t)1  * 1024 * 1024);    // 25.6 MB : x @ W1
    float* agg1 = (float*)(ws + (size_t)28 * 1024 * 1024);    // 25.6 MB : layer-1 agg / h
    float* hw2  = (float*)(ws + (size_t)55 * 1024 * 1024);    // 12.8 MB : h @ W2

    const int TB = 256;

    // degree + symmetric norm
    init_deg_kernel<<<(N_NODES + TB - 1) / TB, TB, 0, stream>>>(dinv, N_NODES);
    count_deg_kernel<<<(N_EDGES + TB - 1) / TB, TB, 0, stream>>>(cols, dinv, N_EDGES);
    rsqrt_kernel<<<(N_NODES + TB - 1) / TB, TB, 0, stream>>>(dinv, N_NODES);

    // layer 1: xw = x @ W1   (6250 x 4 tiles, 8 waves/block)
    {
        int tiles = (N_NODES / 16) * (HID_C / 16);
        gemm_bf16_wmma<IN_C, HID_C><<<(tiles + 7) / 8, TB, 0, stream>>>(x, W1, xw, N_NODES);
    }
    init_agg_kernel<HID_C><<<(N_NODES * HID_C + TB - 1) / TB, TB, 0, stream>>>(
        xw, dinv, b1, agg1, N_NODES);
    {
        long long work = (long long)N_EDGES * HID_C;
        edge_agg_kernel<HID_C><<<(unsigned)((work + TB - 1) / TB), TB, 0, stream>>>(
            xw, rows, cols, dinv, agg1, N_EDGES);
    }
    relu_kernel<<<(N_NODES * HID_C + TB - 1) / TB, TB, 0, stream>>>(agg1, N_NODES * HID_C);

    // layer 2: hw2 = h @ W2
    {
        int tiles = (N_NODES / 16) * (OUT_C / 16);
        gemm_bf16_wmma<HID_C, OUT_C><<<(tiles + 7) / 8, TB, 0, stream>>>(agg1, W2, hw2, N_NODES);
    }
    init_agg_kernel<OUT_C><<<(N_NODES * OUT_C + TB - 1) / TB, TB, 0, stream>>>(
        hw2, dinv, b2, out, N_NODES);
    {
        long long work = (long long)N_EDGES * OUT_C;
        edge_agg_kernel<OUT_C><<<(unsigned)((work + TB - 1) / TB), TB, 0, stream>>>(
            hw2, rows, cols, dinv, out, N_EDGES);
    }
}
